// GAE_47339129537012
// MI455X (gfx1250) — matrix operand order repaired and checked by reading, not compiled, not run on hardware
//
#include <hip/hip_runtime.h>

typedef _Float16 half8  __attribute__((ext_vector_type(8)));
typedef _Float16 half16 __attribute__((ext_vector_type(16)));
typedef float    float8 __attribute__((ext_vector_type(8)));

static constexpr int IN_C  = 128;
static constexpr int HID_C = 64;
static constexpr int OUT_C = 32;

// ---------------- degree / symmetric norm ----------------
__global__ void k_deg_init(float* deg, int n) {
  int i = blockIdx.x * blockDim.x + threadIdx.x;
  if (i < n) deg[i] = 1.0f;                      // self loop contributes 1
}
__global__ void k_deg_accum(const int* __restrict__ dst, float* deg, int e) {
  int i = blockIdx.x * blockDim.x + threadIdx.x;
  if (i < e) atomicAdd(&deg[dst[i]], 1.0f);
}
__global__ void k_rsqrt(float* deg, int n) {
  int i = blockIdx.x * blockDim.x + threadIdx.x;
  if (i < n) deg[i] = rsqrtf(deg[i]);            // deg >= 1 always
}

// ---------------- fp32 -> fp16 helpers ----------------
__global__ void k_f2h(const float* __restrict__ s, _Float16* __restrict__ d, int n) {
  int i = blockIdx.x * blockDim.x + threadIdx.x;
  if (i < n) d[i] = (_Float16)s[i];
}
// W [K x N] f32 row-major  ->  Wt [N x K] f16 (so B fragments load contiguously)
__global__ void k_w_transpose(const float* __restrict__ W, _Float16* __restrict__ Wt,
                              int K, int N) {
  int i = blockIdx.x * blockDim.x + threadIdx.x;
  if (i < K * N) {
    int k = i / N, n = i - k * N;
    Wt[n * K + k] = (_Float16)W[i];
  }
}

// ---------------- WMMA GEMM: C[MxN] = A[MxK] * B[KxN], Bt stored [NxK] ----------------
// One wave32 per 16x16 C tile; K-loop in steps of 32 with v_wmma_f32_16x16x32_f16.
template<int K, int N>
__global__ void k_wmma_gemm(const _Float16* __restrict__ A,
                            const _Float16* __restrict__ Bt,
                            float* __restrict__ C, int M) {
  const int lane  = threadIdx.x & 31;
  const int wave  = (int)((blockIdx.x * (unsigned)blockDim.x + threadIdx.x) >> 5);
  constexpr int NT = N / 16;
  const int mTile = wave / NT;
  const int nTile = wave - mTile * NT;
  if (mTile * 16 >= M) return;                   // wave-uniform: EXEC stays all-ones
  const int g = lane >> 4;                       // half-wave group
  const int c = lane & 15;
  const _Float16* aPtr = A  + (size_t)(mTile * 16 + c) * K;
  const _Float16* bPtr = Bt + (size_t)(nTile * 16 + c) * K;
  float8 acc = {};
  #pragma unroll
  for (int kk = 0; kk < K; kk += 32) {
    // A fragment: ISA layout -> j<8: K=kk+8g+j ; j>=8: K=kk+16+8g+(j-8)
    half8 alo = *(const half8*)(aPtr + kk + 8 * g);
    half8 ahi = *(const half8*)(aPtr + kk + 16 + 8 * g);
    half16 a;
    #pragma unroll
    for (int j = 0; j < 8; ++j) { a[j] = alo[j]; a[j + 8] = ahi[j]; }
    // B fragment from transposed W: j<8: K=kk+8g+j ; j>=8: K=kk+16+8g+(j-8), 16B aligned
    half16 b = __builtin_shufflevector(*(const half8*)(bPtr + kk + 8 * g), *(const half8*)(bPtr + kk + 16 + 8 * g), 0, 1, 2, 3, 4, 5, 6, 7, 8, 9, 10, 11, 12, 13, 14, 15);
    acc = __builtin_amdgcn_wmma_f32_16x16x32_f16(
        /*neg_a=*/false, a, /*neg_b=*/false, b,
        /*c_mod=*/(short)0, acc, /*reuse_a=*/false, /*reuse_b=*/false);
  }
  // C/D layout: VGPR r -> M = r + 8g, N = c
  float* cPtr = C + (size_t)(mTile * 16 + g * 8) * N + nTile * 16 + c;
  #pragma unroll
  for (int r = 0; r < 8; ++r) cPtr[(size_t)r * N] = acc[r];
}

// ---------------- graph aggregation ----------------
template<int LOGF>
__global__ void k_self_init(const float* __restrict__ xw, const float* __restrict__ dinv,
                            float* __restrict__ agg, int n) {
  int t = blockIdx.x * blockDim.x + threadIdx.x;
  if (t < (n << LOGF)) {
    int i = t >> LOGF;
    float di = dinv[i];
    agg[t] = xw[t] * di * di;                    // self-loop message (assignment)
  }
}
template<int LOGF>
__global__ void k_edge_scatter(const int* __restrict__ src, const int* __restrict__ dst,
                               const float* __restrict__ xw, const float* __restrict__ dinv,
                               float* __restrict__ agg, int e) {
  long long t = (long long)blockIdx.x * blockDim.x + threadIdx.x;
  if (t < ((long long)e << LOGF)) {
    int eid = (int)(t >> LOGF);
    int f   = (int)t & ((1 << LOGF) - 1);
    int s = src[eid], d = dst[eid];
    float nrm = dinv[s] * dinv[d];
    atomicAdd(&agg[((size_t)d << LOGF) + f], xw[((size_t)s << LOGF) + f] * nrm);
  }
}
__global__ void k_bias_relu_h(const float* __restrict__ agg, const float* __restrict__ b,
                              _Float16* __restrict__ h, int n) {        // F = 64
  int t = blockIdx.x * blockDim.x + threadIdx.x;
  if (t < n * HID_C) {
    float v = agg[t] + b[t & (HID_C - 1)];
    h[t] = (_Float16)(v > 0.0f ? v : 0.0f);
  }
}
__global__ void k_bias_z(const float* __restrict__ agg, const float* __restrict__ b,
                         float* __restrict__ z, int n) {                // F = 32
  int t = blockIdx.x * blockDim.x + threadIdx.x;
  if (t < n * OUT_C) z[t] = agg[t] + b[t & (OUT_C - 1)];
}

// ---------------- decode: logits[e] = dot(z[src], z[dst]) over 32 ----------------
__global__ void k_decode(const int* __restrict__ s, const int* __restrict__ d,
                         const float* __restrict__ z, float* __restrict__ out, int e) {
  int i = blockIdx.x * blockDim.x + threadIdx.x;
  if (i < e) {
    const float4* zs = (const float4*)(z + (size_t)s[i] * OUT_C);
    const float4* zd = (const float4*)(z + (size_t)d[i] * OUT_C);
    float acc = 0.0f;
    #pragma unroll
    for (int j = 0; j < OUT_C / 4; ++j) {
      float4 a = zs[j], b = zd[j];
      acc += a.x * b.x + a.y * b.y + a.z * b.z + a.w * b.w;
    }
    out[i] = acc;
  }
}

extern "C" void kernel_launch(void* const* d_in, const int* in_sizes, int n_in,
                              void* d_out, int out_size, void* d_ws, size_t ws_size,
                              hipStream_t stream) {
  const float* x  = (const float*)d_in[0];
  const float* W1 = (const float*)d_in[1];
  const float* b1 = (const float*)d_in[2];
  const float* W2 = (const float*)d_in[3];
  const float* b2 = (const float*)d_in[4];
  const int*  ei  = (const int*)d_in[5];
  const int*  dei = (const int*)d_in[6];

  const int N  = in_sizes[0] / IN_C;     // 50000 (multiple of 16)
  const int E  = in_sizes[5] / 2;        // 800000
  const int ED = in_sizes[6] / 2;        // 1600000
  const int* src = ei;   const int* dst = ei + E;
  const int* dsE = dei;  const int* ddE = dei + ED;

  // carve workspace (256B aligned regions)
  char* p = (char*)d_ws;
  auto carve = [&](size_t bytes) { char* r = p; p += (bytes + 255) & ~(size_t)255; return r; };
  float*    dinv = (float*)   carve((size_t)N * 4);
  _Float16* xh   = (_Float16*)carve((size_t)N * IN_C * 2);
  _Float16* w1t  = (_Float16*)carve((size_t)IN_C * HID_C * 2);
  _Float16* w2t  = (_Float16*)carve((size_t)HID_C * OUT_C * 2);
  float*    xw1  = (float*)   carve((size_t)N * HID_C * 4);
  float*    agg1 = (float*)   carve((size_t)N * HID_C * 4);
  _Float16* hh   = (_Float16*)carve((size_t)N * HID_C * 2);
  float*    xw2  = (float*)   carve((size_t)N * OUT_C * 4);
  float*    agg2 = (float*)   carve((size_t)N * OUT_C * 4);
  float*    z    = (float*)   carve((size_t)N * OUT_C * 4);

  const int B = 256;
  auto grid = [&](long long n) { return dim3((unsigned)((n + B - 1) / B)); };

  // normalization
  k_deg_init <<<grid(N), B, 0, stream>>>(dinv, N);
  k_deg_accum<<<grid(E), B, 0, stream>>>(dst, dinv, E);
  k_rsqrt    <<<grid(N), B, 0, stream>>>(dinv, N);

  // precision conversion for WMMA
  k_f2h        <<<grid((long long)N * IN_C), B, 0, stream>>>(x, xh, N * IN_C);
  k_w_transpose<<<grid(IN_C * HID_C), B, 0, stream>>>(W1, w1t, IN_C, HID_C);
  k_w_transpose<<<grid(HID_C * OUT_C), B, 0, stream>>>(W2, w2t, HID_C, OUT_C);

  // layer 1: GEMM (WMMA) -> self loops -> edge scatter -> bias+ReLU (f16 out)
  {
    long long waves = (long long)(N / 16) * (HID_C / 16);
    k_wmma_gemm<IN_C, HID_C><<<grid(waves * 32), B, 0, stream>>>(xh, w1t, xw1, N);
  }
  k_self_init<6>   <<<grid((long long)N * HID_C), B, 0, stream>>>(xw1, dinv, agg1, N);
  k_edge_scatter<6><<<grid((long long)E * HID_C), B, 0, stream>>>(src, dst, xw1, dinv, agg1, E);
  k_bias_relu_h    <<<grid((long long)N * HID_C), B, 0, stream>>>(agg1, b1, hh, N);

  // layer 2
  {
    long long waves = (long long)(N / 16) * (OUT_C / 16);
    k_wmma_gemm<HID_C, OUT_C><<<grid(waves * 32), B, 0, stream>>>(hh, w2t, xw2, N);
  }
  k_self_init<5>   <<<grid((long long)N * OUT_C), B, 0, stream>>>(xw2, dinv, agg2, N);
  k_edge_scatter<5><<<grid((long long)E * OUT_C), B, 0, stream>>>(src, dst, xw2, dinv, agg2, E);
  k_bias_z         <<<grid((long long)N * OUT_C), B, 0, stream>>>(agg2, b2, z, N);

  // decode
  k_decode<<<grid(ED), B, 0, stream>>>(dsE, ddE, z, (float*)d_out, ED);
}
